// QuanvolutionGen_65481071407835
// MI455X (gfx1250) — compile-verified, matrix operand order checked by reading, not run on hardware
//
#include <hip/hip_runtime.h>
#include <cstddef>

typedef __attribute__((ext_vector_type(8)))  _Float16 v8h;
typedef __attribute__((ext_vector_type(16))) _Float16 v16h;
typedef __attribute__((ext_vector_type(8)))  float    v8f;

#define DEVINL __device__ __forceinline__

// ---------------- 4-qubit gate helpers (state in registers) ----------------
// Wire w acts on bit (3-w) of the flat 16-dim index; stride S = 8 >> w.

template<int S> DEVINL void gate_ry(float* re, float* im, float ct, float st) {
#pragma unroll
  for (int i = 0; i < 16; ++i) if ((i & S) == 0) {
    const int j = i | S;
    float ri = re[i], rj = re[j], ii = im[i], ij = im[j];
    re[i] = ct*ri - st*rj;  re[j] = st*ri + ct*rj;
    im[i] = ct*ii - st*ij;  im[j] = st*ii + ct*ij;
  }
}

template<int S> DEVINL void gate_rx(float* re, float* im, float ct, float st) {
#pragma unroll
  for (int i = 0; i < 16; ++i) if ((i & S) == 0) {
    const int j = i | S;
    float ri = re[i], rj = re[j], ii = im[i], ij = im[j];
    re[i] = ct*ri + st*ij;  im[i] = ct*ii - st*rj;   // v_i' = ct*v_i - i*st*v_j
    re[j] = ct*rj + st*ii;  im[j] = ct*ij - st*ri;   // v_j' = -i*st*v_i + ct*v_j
  }
}

template<int S> DEVINL void gate_rz(float* re, float* im, float ct, float st) {
#pragma unroll
  for (int i = 0; i < 16; ++i) {
    float r = re[i], q = im[i];
    if (i & S) { re[i] = ct*r - st*q; im[i] = ct*q + st*r; }  // e^{+i t/2}
    else       { re[i] = ct*r + st*q; im[i] = ct*q - st*r; }  // e^{-i t/2}
  }
}

template<int SC, int ST> DEVINL void gate_cnot(float* re, float* im) {
#pragma unroll
  for (int i = 0; i < 16; ++i) if ((i & SC) && !(i & ST)) {
    const int j = i | ST;
    float t;
    t = re[i]; re[i] = re[j]; re[j] = t;
    t = im[i]; im[i] = im[j]; im[j] = t;
  }
}

DEVINL void cnot_ring(float* re, float* im) {
  gate_cnot<8,4>(re, im);   // CNOT(0,1)
  gate_cnot<4,2>(re, im);   // CNOT(1,2)
  gate_cnot<2,1>(re, im);   // CNOT(2,3)
  gate_cnot<1,8>(re, im);   // CNOT(3,0)
}

// ---------------- Kernel 1: per-patch circuit simulation -> f16 features ---
__global__ void __launch_bounds__(256)
quanv_features(const float* __restrict__ x, const float* __restrict__ theta,
               _Float16* __restrict__ feat, int nCirc) {
  int idx = blockIdx.x * 256 + threadIdx.x;
  if (idx >= nCirc) return;
  int b = idx / 196, p = idx - b * 196;
  int prow = p / 14, pcol = p - prow * 14;
  const float* xp = x + (size_t)b * 784 + prow * 56 + pcol * 2;
  float p0 = xp[0], p1 = xp[1], p2 = xp[28], p3 = xp[29];

  float ct[12], st[12];
#pragma unroll
  for (int k = 0; k < 12; ++k) {
    float t = 0.5f * theta[k];   // uniform scalar loads
    ct[k] = cosf(t); st[k] = sinf(t);
  }

  // Encoder RY(p_w) on |0000>: product state, purely real.
  float c0 = cosf(0.5f * p0), s0 = sinf(0.5f * p0);
  float c1 = cosf(0.5f * p1), s1 = sinf(0.5f * p1);
  float c2 = cosf(0.5f * p2), s2 = sinf(0.5f * p2);
  float c3 = cosf(0.5f * p3), s3 = sinf(0.5f * p3);

  float re[16], im[16];
#pragma unroll
  for (int i = 0; i < 16; ++i) {
    re[i] = ((i & 8) ? s0 : c0) * ((i & 4) ? s1 : c1) *
            ((i & 2) ? s2 : c2) * ((i & 1) ? s3 : c3);
    im[i] = 0.f;
  }

  // 12-op layer: rots=[RX,RY,RZ], gate k -> rots[k%3] on wire k%4; ring after k%4==3
  gate_rx<8>(re, im, ct[0],  st[0]);
  gate_ry<4>(re, im, ct[1],  st[1]);
  gate_rz<2>(re, im, ct[2],  st[2]);
  gate_rx<1>(re, im, ct[3],  st[3]);
  cnot_ring(re, im);
  gate_ry<8>(re, im, ct[4],  st[4]);
  gate_rz<4>(re, im, ct[5],  st[5]);
  gate_rx<2>(re, im, ct[6],  st[6]);
  gate_ry<1>(re, im, ct[7],  st[7]);
  cnot_ring(re, im);
  gate_rz<8>(re, im, ct[8],  st[8]);
  gate_rx<4>(re, im, ct[9],  st[9]);
  gate_ry<2>(re, im, ct[10], st[10]);
  gate_rz<1>(re, im, ct[11], st[11]);
  cnot_ring(re, im);

  float m[4] = {0.f, 0.f, 0.f, 0.f};
#pragma unroll
  for (int i = 0; i < 16; ++i) {
    float pb = re[i]*re[i] + im[i]*im[i];
    m[0] += (i & 8) ? -pb : pb;
    m[1] += (i & 4) ? -pb : pb;
    m[2] += (i & 2) ? -pb : pb;
    m[3] += (i & 1) ? -pb : pb;
  }

  v8h f;
#pragma unroll
  for (int w = 0; w < 4; ++w) {
    f[w]     = (_Float16)fabsf(m[w]);
    f[4 + w] = (_Float16)(m[w] > 0.f ? 1.f : (m[w] < 0.f ? -1.f : 0.f));
  }
  *reinterpret_cast<v8h*>(feat + (size_t)idx * 8) = f;   // 16B packed store
}

// ---------------- Kernel 2: pad W (10x1568 f32) -> 16x1568 f16 -------------
__global__ void __launch_bounds__(256)
convert_w(const float* __restrict__ W, _Float16* __restrict__ w16) {
  int idx = blockIdx.x * 256 + threadIdx.x;
  if (idx >= 16 * 1568) return;
  int n = idx / 1568, k = idx - n * 1568;
  w16[idx] = (n < 10) ? (_Float16)W[n * 1568 + k] : (_Float16)0.f;
}

// ---------------- Kernel 3: WMMA GEMM + bias + log_softmax -----------------
// One block per 16-batch M tile; 8 waves split the 49 K-chunks of 32.
__global__ void __launch_bounds__(256)
gemm_logsoftmax(const _Float16* __restrict__ feat, const _Float16* __restrict__ w16,
                const float* __restrict__ bias, float* __restrict__ out) {
  __shared__ float Ct[16][17];
  const int tid = threadIdx.x;
  for (int i = tid; i < 16 * 17; i += 256) (&Ct[0][0])[i] = 0.f;
  __syncthreads();

  const int wave = tid >> 5;
  const int lane = tid & 31;
  const int mn   = lane & 15;     // M row for A, N col for B/C
  const int half = lane >> 4;     // lanes 16-31 hold K+8 (16-bit operand layout)
  const int tileM = blockIdx.x * 16;

  v8f acc = {};
  for (int kc = wave; kc < 49; kc += 8) {
    const int k0 = kc * 32;
    // A fragment: row (tileM+mn), K = k0+half*8..+7 and k0+16+half*8..+7
    const _Float16* arow = feat + (size_t)(tileM + mn) * 1568 + k0 + half * 8;
    v8h alo = *reinterpret_cast<const v8h*>(arow);
    v8h ahi = *reinterpret_cast<const v8h*>(arow + 16);
    v16h a = __builtin_shufflevector(alo, ahi, 0,1,2,3,4,5,6,7,8,9,10,11,12,13,14,15);
    // B fragment: col mn of W^T == row mn of padded W
    const _Float16* brow = w16 + (size_t)mn * 1568 + k0 + half * 8;
    v8h blo = *reinterpret_cast<const v8h*>(brow);
    v8h bhi = *reinterpret_cast<const v8h*>(brow + 16);
    v16h bb = __builtin_shufflevector(blo, bhi, 0,1,2,3,4,5,6,7,8,9,10,11,12,13,14,15);

    acc = __builtin_amdgcn_wmma_f32_16x16x32_f16(
        /*neg_a=*/false, a, /*neg_b=*/false, bb,
        /*c_mod=*/(short)0, acc, /*reuse_a=*/false, /*reuse_b=*/false);
  }

  // Cross-wave reduction of partial C tiles via LDS float atomics (ds_add_f32).
#pragma unroll
  for (int r = 0; r < 8; ++r)
    atomicAdd(&Ct[r + 8 * half][mn], acc[r]);
  __syncthreads();

  if (tid < 16) {
    float l[10];
    float mx = -3.0e38f;
#pragma unroll
    for (int n = 0; n < 10; ++n) { l[n] = Ct[tid][n] + bias[n]; mx = fmaxf(mx, l[n]); }
    float s = 0.f;
#pragma unroll
    for (int n = 0; n < 10; ++n) s += expf(l[n] - mx);
    float lse = mx + logf(s);
    float* o = out + (size_t)(tileM + tid) * 10;
#pragma unroll
    for (int n = 0; n < 10; ++n) o[n] = l[n] - lse;
  }
}

// ---------------- Launcher -------------------------------------------------
extern "C" void kernel_launch(void* const* d_in, const int* in_sizes, int n_in,
                              void* d_out, int out_size, void* d_ws, size_t ws_size,
                              hipStream_t stream) {
  const float* x     = (const float*)d_in[0];   // (B,28,28) f32
  const float* theta = (const float*)d_in[1];   // (12,)     f32
  const float* W     = (const float*)d_in[2];   // (10,1568) f32
  const float* bias  = (const float*)d_in[3];   // (10,)     f32
  float* out = (float*)d_out;                   // (B,10)    f32 log-probs

  const int nB    = in_sizes[0] / (28 * 28);    // 2048
  const int nCirc = nB * 196;

  // Workspace: feat f16 (B*1568) then padded W f16 (16*1568). ~6.5 MB total.
  _Float16* feat = (_Float16*)d_ws;
  _Float16* w16  = (_Float16*)((char*)d_ws + (size_t)nB * 1568 * sizeof(_Float16));

  quanv_features<<<(nCirc + 255) / 256, 256, 0, stream>>>(x, theta, feat, nCirc);
  convert_w<<<(16 * 1568 + 255) / 256, 256, 0, stream>>>(W, w16);
  gemm_logsoftmax<<<nB / 16, 256, 0, stream>>>(feat, w16, bias, out);
}
